// LocalSelfAttention_24361054503216
// MI455X (gfx1250) — matrix operand order repaired and checked by reading, not compiled, not run on hardware
//
#include <hip/hip_runtime.h>

// ---------------------------------------------------------------------------
// Types for CDNA5 WMMA (wave32, 16x16x32 f16 -> f32)
// ---------------------------------------------------------------------------
typedef _Float16     half_t;
typedef _Float16     v16h __attribute__((ext_vector_type(16)));
typedef float        v8f  __attribute__((ext_vector_type(8)));
typedef unsigned int v4u  __attribute__((ext_vector_type(4)));
typedef float        v4f  __attribute__((ext_vector_type(4)));

union FragA  { v16h v; v4u q[2]; };
union Pack8h { half_t h[8]; v4u q; };

#define WMMA_F16(a, b, c) \
    __builtin_amdgcn_wmma_f32_16x16x32_f16(false, (a), false, (b), (short)0, (c), false, false)

// Problem constants
#define BATCH 2
#define SEQ   8192
#define HID   1024
#define NHEAD 16
#define DHEAD 64
#define CLEN  128
#define NCHNK 64          // SEQ / CLEN
#define NTOK  (BATCH*SEQ) // 16384

// ---------------------------------------------------------------------------
// Kernel 0a: convert hidden_states fp32 -> f16, same row-major [t][k] layout.
// ---------------------------------------------------------------------------
__global__ __launch_bounds__(256)
void cvt_x(const float* __restrict__ X, half_t* __restrict__ Xh)
{
    const size_t i = ((size_t)blockIdx.x * 256 + threadIdx.x) * 8;
    v4f a = *(const v4f*)(X + i);
    v4f b = *(const v4f*)(X + i + 4);
    Pack8h p;
    p.h[0] = (half_t)a.x; p.h[1] = (half_t)a.y; p.h[2] = (half_t)a.z; p.h[3] = (half_t)a.w;
    p.h[4] = (half_t)b.x; p.h[5] = (half_t)b.y; p.h[6] = (half_t)b.z; p.h[7] = (half_t)b.w;
    *(v4u*)(Xh + i) = p.q;
}

// ---------------------------------------------------------------------------
// Kernel 0b: convert + transpose weights fp32 [k][n] -> f16 Wt [mode][n][k],
// folding the 1/sqrt(DH)=0.125 key scale into Wk (mode 1).
// ---------------------------------------------------------------------------
__global__ __launch_bounds__(256)
void cvt_w(const float* __restrict__ Wq,
           const float* __restrict__ Wk,
           const float* __restrict__ Wv,
           half_t* __restrict__ Wt)
{
    const int mode = blockIdx.z;
    const float* W = (mode == 0) ? Wq : ((mode == 1) ? Wk : Wv);
    const float scale = (mode == 1) ? 0.125f : 1.0f;

    const int k0 = blockIdx.x * 32;
    const int n0 = blockIdx.y * 256;
    const int tid = threadIdx.x;

    __shared__ float Ws[32][256];
#pragma unroll
    for (int i = 0; i < 32; ++i)
        Ws[i][tid] = W[(size_t)(k0 + i) * HID + n0 + tid];
    __syncthreads();

    half_t* out = Wt + (size_t)mode * HID * HID + (size_t)(n0 + tid) * HID + k0;
#pragma unroll
    for (int j = 0; j < 32; j += 8) {
        Pack8h p;
#pragma unroll
        for (int e = 0; e < 8; ++e)
            p.h[e] = (half_t)(Ws[j + e][tid] * scale);
        *(v4u*)(out + j) = p.q;
    }
}

// ---------------------------------------------------------------------------
// Kernel A: QKV projection GEMM, LDS-free, register double-buffered.
//   z = 0: Q  [B,NH,S,64]
//   z = 1: K  [B,NH,S,64]   (scale pre-folded into Wk)
//   z = 2: Vt [B,NH,64,S]   (transposed store for PV B-fragments)
// Per block: 128(M) x 128(N) tile, 8 waves x (16 rows x 128 cols), K-step 32.
// ---------------------------------------------------------------------------
__global__ __launch_bounds__(256)
void qkv_gemm(const half_t* __restrict__ Xh,
              const half_t* __restrict__ Wt,
              half_t* __restrict__ Qf,
              half_t* __restrict__ Kf,
              half_t* __restrict__ Vf)
{
    const int mode = blockIdx.z;
    half_t* out = (mode == 0) ? Qf : ((mode == 1) ? Kf : Vf);
    const half_t* Wm = Wt + (size_t)mode * HID * HID;

    const int m0 = blockIdx.x * 128;   // token tile
    const int n0 = blockIdx.y * 128;   // output-feature tile

    const int tid  = threadIdx.x;
    const int wave = tid >> 5;
    const int lane = tid & 31;
    const int lr   = lane & 15;
    const int lhi  = lane >> 4;

    v8f acc[8];
    const v8f vzero = {0.f,0.f,0.f,0.f,0.f,0.f,0.f,0.f};
#pragma unroll
    for (int i = 0; i < 8; ++i) acc[i] = vzero;

    // per-lane base pointers (k advances by immediate offsets)
    const half_t* xp = Xh + (size_t)(m0 + wave * 16 + lr) * HID + lhi * 16;
    const half_t* wp = Wm + (size_t)(n0 + lr) * HID + lhi * 16;

    // ---- software pipeline: ping-pong fragment buffers in registers
    FragA a[2];
    FragA bf[2][8];

    a[0].q[0] = *(const v4u*)(xp);
    a[0].q[1] = *(const v4u*)(xp + 8);
#pragma unroll
    for (int nt = 0; nt < 8; ++nt) {
        const half_t* bp = wp + (size_t)nt * 16 * HID;
        bf[0][nt].q[0] = *(const v4u*)(bp);
        bf[0][nt].q[1] = *(const v4u*)(bp + 8);
    }

#pragma unroll 2                       // makes cur/nxt compile-time constant
    for (int kk = 0; kk < HID; kk += 32) {
        const int cur = (kk >> 5) & 1;
        const int nxt = cur ^ 1;
        const int kn  = (kk + 32) & (HID - 1);   // last iter wraps to 0 (discarded)

        // prefetch next k-step while current WMMAs run
        a[nxt].q[0] = *(const v4u*)(xp + kn);
        a[nxt].q[1] = *(const v4u*)(xp + kn + 8);
#pragma unroll
        for (int nt = 0; nt < 8; ++nt) {
            const half_t* bp = wp + (size_t)nt * 16 * HID + kn;
            bf[nxt][nt].q[0] = *(const v4u*)(bp);
            bf[nxt][nt].q[1] = *(const v4u*)(bp + 8);
        }

#pragma unroll
        for (int nt = 0; nt < 8; ++nt)
            acc[nt] = WMMA_F16(a[cur].v, bf[cur][nt].v, acc[nt]);
    }

    // ---- store f16: C-tile layout VGPR r -> row (r + 8*lhi), col = lr
    const int rofs = lhi * 8;
#pragma unroll
    for (int nt = 0; nt < 8; ++nt) {
        const int n = n0 + nt * 16 + lr;
        const int h = n >> 6;
        const int d = n & 63;
#pragma unroll
        for (int r = 0; r < 8; ++r) {
            const int t = m0 + wave * 16 + rofs + r;  // global token
            const int b = t >> 13;
            const int s = t & (SEQ - 1);
            size_t idx;
            if (mode == 2) idx = (((size_t)(b * NHEAD + h)) * DHEAD + d) * SEQ + s;
            else           idx = (((size_t)(b * NHEAD + h)) * SEQ + s) * DHEAD + d;
            out[idx] = (half_t)acc[nt][r];
        }
    }
}

// ---------------------------------------------------------------------------
// Kernel B: chunked local attention, flash-style online softmax.
// grid = (NC, NH, B); block = 128 (4 independent waves of 32 query rows).
// Window = 256 keys: previous chunk + current chunk (NB=1, NA=0).
// ---------------------------------------------------------------------------
__global__ __launch_bounds__(128)
void local_attn(const half_t* __restrict__ Qf,
                const half_t* __restrict__ Kf,
                const half_t* __restrict__ Vf,
                float* __restrict__ out)
{
    const int chunk = blockIdx.x;           // 0..63
    const int h     = blockIdx.y;           // 0..15
    const int b     = blockIdx.z;           // 0..1
    const int bh    = b * NHEAD + h;
    const int wave  = threadIdx.x >> 5;
    const int lane  = threadIdx.x & 31;
    const int lr    = lane & 15;
    const int lhi   = lane >> 4;

    // per-wave P chunk (32 rows x 32 keys f16), stride 40 halves -> b128 A-frags
    __shared__ __align__(16) half_t Pl[4][32 * 40];
    half_t* pw = Pl[wave];

    const int    m0       = wave * 32;                    // query rows of this wave
    const size_t qkvBase  = (size_t)bh * SEQ * DHEAD;
    const int    prev     = (chunk + NCHNK - 1) & (NCHNK - 1);

    // ---- Q A-fragments: 2 M-tiles x 2 k-steps, kept in registers
    FragA qfrag[2][2];
#pragma unroll
    for (int t = 0; t < 2; ++t) {
        const int qrow = chunk * CLEN + m0 + t * 16 + lr;
        const half_t* qp = Qf + qkvBase + (size_t)qrow * DHEAD;
#pragma unroll
        for (int ks = 0; ks < 2; ++ks) {
            const int kb = ks * 32 + lhi * 16;
            qfrag[t][ks].q[0] = *(const v4u*)(qp + kb);
            qfrag[t][ks].q[1] = *(const v4u*)(qp + kb + 8);
        }
    }

    const v8f vzero = {0.f,0.f,0.f,0.f,0.f,0.f,0.f,0.f};
    v8f o[2][4];
#pragma unroll
    for (int t = 0; t < 2; ++t)
#pragma unroll
        for (int dt = 0; dt < 4; ++dt) o[t][dt] = vzero;

    float mrow[2][8], lsum[2][8];
#pragma unroll
    for (int t = 0; t < 2; ++t)
#pragma unroll
        for (int r = 0; r < 8; ++r) { mrow[t][r] = -1e30f; lsum[t][r] = 0.f; }

    // ======== stream over 8 key-chunks of 32 keys ========
    for (int kc = 0; kc < 8; ++kc) {
        const int j0       = kc * 32;                                  // window offset
        const int srcChunk = (j0 < CLEN) ? prev : chunk;
        const int keybase  = srcChunk * CLEN + ((j0 < CLEN) ? j0 : (j0 - CLEN));

        // ---- prefetch V B-fragments NOW; latency hides under S + softmax
        FragA vfr[4];
#pragma unroll
        for (int dt = 0; dt < 4; ++dt) {
            const half_t* vp = Vf + qkvBase + (size_t)(dt * 16 + lr) * SEQ
                                 + keybase + lhi * 16;
            vfr[dt].q[0] = *(const v4u*)(vp);
            vfr[dt].q[1] = *(const v4u*)(vp + 8);
        }

        // ---- S = Q K^T for this 32-key chunk: 2(M) x 2(N) C-tiles
        v8f s[2][2];
#pragma unroll
        for (int t = 0; t < 2; ++t) { s[t][0] = vzero; s[t][1] = vzero; }

#pragma unroll
        for (int nt = 0; nt < 2; ++nt) {
            const int key = keybase + nt * 16 + lr;
            const half_t* kp = Kf + qkvBase + (size_t)key * DHEAD;
            FragA kfr[2];
#pragma unroll
            for (int ks = 0; ks < 2; ++ks) {
                const int db = ks * 32 + lhi * 16;
                kfr[ks].q[0] = *(const v4u*)(kp + db);
                kfr[ks].q[1] = *(const v4u*)(kp + db + 8);
            }
#pragma unroll
            for (int ks = 0; ks < 2; ++ks)
#pragma unroll
                for (int t = 0; t < 2; ++t)
                    s[t][nt] = WMMA_F16(qfrag[t][ks].v, kfr[ks].v, s[t][nt]);
        }

        // ---- causal mask (global positions; wrapped chunk-0 keys auto-masked)
#pragma unroll
        for (int t = 0; t < 2; ++t) {
            const int qpos0 = chunk * CLEN + m0 + t * 16 + lhi * 8;
#pragma unroll
            for (int nt = 0; nt < 2; ++nt) {
                const int kvpos = keybase + nt * 16 + lr;
#pragma unroll
                for (int r = 0; r < 8; ++r) {
                    const float v = s[t][nt][r];
                    s[t][nt][r] = ((qpos0 + r) >= kvpos) ? v : -1e9f;
                }
            }
        }

        // ---- online softmax update (row reduce across 16-lane halves)
        float scalef[2][8];
#pragma unroll
        for (int t = 0; t < 2; ++t) {
#pragma unroll
            for (int r = 0; r < 8; ++r) {
                float mx = fmaxf(s[t][0][r], s[t][1][r]);
#pragma unroll
                for (int d = 1; d < 16; d <<= 1)
                    mx = fmaxf(mx, __shfl_xor(mx, d, 32));
                const float mn = fmaxf(mrow[t][r], mx);
                scalef[t][r] = __expf(mrow[t][r] - mn);
                mrow[t][r] = mn;

                const float p0 = __expf(s[t][0][r] - mn);
                const float p1 = __expf(s[t][1][r] - mn);
                s[t][0][r] = p0;
                s[t][1][r] = p1;
                float sm = p0 + p1;
#pragma unroll
                for (int d = 1; d < 16; d <<= 1)
                    sm += __shfl_xor(sm, d, 32);
                lsum[t][r] = lsum[t][r] * scalef[t][r] + sm;
#pragma unroll
                for (int dt = 0; dt < 4; ++dt)
                    o[t][dt][r] *= scalef[t][r];
            }
        }

        // ---- P (f32 C-layout) -> LDS f16 row-major [row][key]
#pragma unroll
        for (int t = 0; t < 2; ++t)
#pragma unroll
            for (int nt = 0; nt < 2; ++nt)
#pragma unroll
                for (int r = 0; r < 8; ++r)
                    pw[(t * 16 + lhi * 8 + r) * 40 + nt * 16 + lr] = (half_t)s[t][nt][r];

        asm volatile("s_wait_dscnt 0" ::: "memory");   // drain LDS stores (wave-local)

        // ---- reload P as A-fragments (k-step = 32 keys)
        FragA pf[2];
#pragma unroll
        for (int t = 0; t < 2; ++t) {
            const half_t* pr = &pw[(t * 16 + lr) * 40 + lhi * 16];
            pf[t].q[0] = *(const v4u*)(pr);
            pf[t].q[1] = *(const v4u*)(pr + 8);
        }

        // ---- O += P * V  (V fragments already resident)
#pragma unroll
        for (int dt = 0; dt < 4; ++dt)
#pragma unroll
            for (int t = 0; t < 2; ++t)
                o[t][dt] = WMMA_F16(pf[t].v, vfr[dt].v, o[t][dt]);
    }

    // ---- epilogue: normalize and store fp32 out [B, S, NH*DH]
#pragma unroll
    for (int t = 0; t < 2; ++t) {
#pragma unroll
        for (int r = 0; r < 8; ++r) {
            const float inv  = 1.0f / lsum[t][r];
            const int   srow = chunk * CLEN + m0 + t * 16 + lhi * 8 + r;
            float* op = out + ((size_t)b * SEQ + srow) * (NHEAD * DHEAD) + h * DHEAD + lr;
#pragma unroll
            for (int dt = 0; dt < 4; ++dt)
                op[dt * 16] = o[t][dt][r] * inv;
        }
    }
}

// ---------------------------------------------------------------------------
extern "C" void kernel_launch(void* const* d_in, const int* in_sizes, int n_in,
                              void* d_out, int out_size, void* d_ws, size_t ws_size,
                              hipStream_t stream)
{
    const float* X  = (const float*)d_in[0];
    const float* Wq = (const float*)d_in[1];
    const float* Wk = (const float*)d_in[2];
    const float* Wv = (const float*)d_in[3];

    // workspace layout (halves): Xh | Wt(3) | Q | K | Vt
    const size_t nX   = (size_t)NTOK * HID;            // 16.7M
    const size_t nW   = (size_t)3 * HID * HID;         // 3.1M
    const size_t nQKV = (size_t)NTOK * (NHEAD * DHEAD);

    half_t* Xh = (half_t*)d_ws;
    half_t* Wt = Xh + nX;
    half_t* Qf = Wt + nW;
    half_t* Kf = Qf + nQKV;
    half_t* Vf = Kf + nQKV;

    cvt_x<<<dim3((unsigned)(nX / (256 * 8))), 256, 0, stream>>>(X, Xh);
    cvt_w<<<dim3(HID / 32, HID / 256, 3), 256, 0, stream>>>(Wq, Wk, Wv, Wt);

    dim3 gA(NTOK / 128, (NHEAD * DHEAD) / 128, 3);   // (128, 8, 3)
    qkv_gemm<<<gA, 256, 0, stream>>>(Xh, Wt, Qf, Kf, Vf);

    dim3 gB(NCHNK, NHEAD, BATCH);                    // (64, 16, 2)
    local_attn<<<gB, 128, 0, stream>>>(Qf, Kf, Vf, (float*)d_out);
}
